// LSTM_81492709474432
// MI455X (gfx1250) — matrix operand order repaired and checked by reading, not compiled, not run on hardware
//
#include <hip/hip_runtime.h>

// ---------------------------------------------------------------------------
// Types for CDNA5 WMMA (wave32): 16x16x32 bf16 -> f32
// ---------------------------------------------------------------------------
typedef __attribute__((ext_vector_type(16))) __bf16 v16bf;
typedef __attribute__((ext_vector_type(8)))  float  v8f;

__device__ __forceinline__ v8f wmma_bf16(v16bf a, v16bf b, v8f c) {
    // (neg_a, A, neg_b, B, c_mod, C, reuse_a, reuse_b)
    return __builtin_amdgcn_wmma_f32_16x16x32_bf16(false, a, false, b,
                                                   (short)0, c, false, false);
}

// A fragment (16x32 bf16) from row-major f32 (convert on the fly).
// Lanes 0-15: rows M=0..15, K 0..15; lanes 16-31: K 16..31 (sequential).
__device__ __forceinline__ v16bf load_a_frag_f32(const float* __restrict__ A,
                                                 int lda, int m0, int k0,
                                                 int lane) {
    int row = m0 + (lane & 15);
    int hi  = (lane >> 4) & 1;
    const float* p = A + (size_t)row * lda + k0;
    v16bf a;
#pragma unroll
    for (int j = 0; j < 8; ++j) a[j]     = (__bf16)p[hi * 16 + j];
#pragma unroll
    for (int j = 0; j < 8; ++j) a[8 + j] = (__bf16)p[hi * 16 + 8 + j];
    return a;
}

// A fragment from a row-major bf16 matrix (no conversion: 2x16B loads).
__device__ __forceinline__ v16bf load_a_frag_bf16(const __bf16* __restrict__ A,
                                                  int lda, int m0, int k0,
                                                  int lane) {
    int row = m0 + (lane & 15);
    int hi  = (lane >> 4) & 1;
    const __bf16* p = A + (size_t)row * lda + k0;
    v16bf a;
#pragma unroll
    for (int j = 0; j < 8; ++j) a[j]     = p[hi * 16 + j];
#pragma unroll
    for (int j = 0; j < 8; ++j) a[8 + j] = p[hi * 16 + 8 + j];
    return a;
}

// B fragment (32x16 bf16) with B[k][n] = W[n][k], W row-major [N,K] bf16.
// Lanes 0-15 hold K=0..15, lanes 16-31 K=16..31 (sequential).
__device__ __forceinline__ v16bf load_b_frag(const __bf16* __restrict__ W,
                                             int ldw, int n0, int k0,
                                             int lane) {
    int col = n0 + (lane & 15);
    int hi  = (lane >> 4) & 1;
    const __bf16* p = W + (size_t)col * ldw + k0 + hi * 16;
    v16bf b;
#pragma unroll
    for (int j = 0; j < 16; ++j) b[j] = p[j];
    return b;
}

// B fragment from LDS, pre-arranged in fragment order:
// whf[((g*16 + kt)*32 + lane)*16 + j]  -> two ds_load_b128 per lane.
__device__ __forceinline__ v16bf load_b_frag_sm(const __bf16* whf, int g,
                                                int kt, int lane) {
    const __bf16* p = whf + ((((g * 16 + kt) * 32) + lane) << 4);
    v16bf b;
#pragma unroll
    for (int j = 0; j < 16; ++j) b[j] = p[j];
    return b;
}

// CDNA5 async global->LDS copy (ASYNCcnt-tracked). lds_off is a raw LDS byte
// offset (dynamic-LDS base is 0 here: no static __shared__ in this kernel).
__device__ __forceinline__ void async_load_b128(unsigned lds_off,
                                                const void* gaddr) {
    asm volatile("global_load_async_to_lds_b128 %0, %1, off"
                 :: "v"(lds_off), "v"(gaddr) : "memory");
}
__device__ __forceinline__ void wait_asynccnt0() {
    asm volatile("s_wait_asynccnt 0x0" ::: "memory");
}

// ---------------------------------------------------------------------------
// Problem constants
// ---------------------------------------------------------------------------
#define TT   2048
#define BB   64
#define HH   512
#define G4H  2048           // 4*H
#define CT   128            // time-chunk length (xg staged per chunk)

// Scan-kernel dynamic LDS layout
#define WHF_BYTES   ((size_t)4 * 16 * 32 * 16 * 2)   // 65536: Wh fragments
#define GATES_BYTES ((size_t)4 * BB * 16 * 4)        // 16384: gate exchange
#define SMEM_BYTES  (WHF_BYTES + GATES_BYTES)

// Workspace layout (bytes)
#define XG_OFF    ((size_t)0)                        // CT*BB*G4H f32
#define XG_BYTES  ((size_t)CT * BB * G4H * 4)
#define WXB_OFF   (XG_OFF + XG_BYTES)                // 2048*512 bf16
#define WB_BYTES  ((size_t)G4H * HH * 2)
#define WHB_OFF   (WXB_OFF + WB_BYTES)
#define HB0_OFF   (WHB_OFF + WB_BYTES)               // 64*512 bf16 ping
#define HB_BYTES  ((size_t)BB * HH * 2)
#define HB1_OFF   (HB0_OFF + HB_BYTES)               // 64*512 bf16 pong
#define HF_OFF    (HB1_OFF + HB_BYTES)               // 64*512 f32 (for h_n)
#define ST_BYTES  ((size_t)BB * HH * 4)
#define C_OFF     (HF_OFF + ST_BYTES)                // 64*512 f32 cell state
#define CNT_OFF   (C_OFF + ST_BYTES)

// ---------------------------------------------------------------------------
// f32 -> bf16 weight conversion
// ---------------------------------------------------------------------------
__global__ void cvt_bf16_kernel(const float* __restrict__ src,
                                __bf16* __restrict__ dst, int n) {
    int i = blockIdx.x * blockDim.x + threadIdx.x;
    if (i < n) dst[i] = (__bf16)src[i];
}

// Reset barrier counter; optionally zero h/c state (start of a layer)
__global__ void prep_kernel(unsigned* __restrict__ cnt,
                            __bf16* __restrict__ h0, __bf16* __restrict__ h1,
                            float* __restrict__ hf, float* __restrict__ c,
                            int zero_state) {
    int i = blockIdx.x * blockDim.x + threadIdx.x;
    if (i == 0) *cnt = 0u;
    if (zero_state) {
        for (int k = i; k < BB * HH; k += gridDim.x * blockDim.x) {
            h0[k] = (__bf16)0.f; h1[k] = (__bf16)0.f;
            hf[k] = 0.f; c[k] = 0.f;
        }
    }
}

// ---------------------------------------------------------------------------
// Input-projection GEMM: Out[m,n] = sum_k A[m,k]*W[n,k] + bx[n] + bh[n]
// A: f32 [Mrows,512] row-major, W: bf16 [2048,512] row-major.
// Block 256 threads = 8 waves; wave covers 16 rows x 64 cols.
// grid = (Mrows/16, 4)
// ---------------------------------------------------------------------------
__global__ __launch_bounds__(256) void gemm_xg_kernel(
    const float* __restrict__ A, int lda,
    const __bf16* __restrict__ Wb,
    const float* __restrict__ bx, const float* __restrict__ bh,
    float* __restrict__ Out) {
    int lane = threadIdx.x & 31;
    int wave = threadIdx.x >> 5;
    int m0 = blockIdx.x * 16;
    int n0 = blockIdx.y * 512 + wave * 64;

    v8f acc[4] = {v8f{}, v8f{}, v8f{}, v8f{}};
    for (int k0 = 0; k0 < HH; k0 += 32) {
        v16bf a = load_a_frag_f32(A, lda, m0, k0, lane);
#pragma unroll
        for (int c = 0; c < 4; ++c) {
            v16bf b = load_b_frag(Wb, HH, n0 + c * 16, k0, lane);
            acc[c] = wmma_bf16(a, b, acc[c]);
        }
    }
    int hi = lane >> 4, jc = lane & 15;
#pragma unroll
    for (int c = 0; c < 4; ++c) {
        int n = n0 + c * 16 + jc;
        float bias = bx[n] + bh[n];
#pragma unroll
        for (int r = 0; r < 8; ++r) {
            int m = m0 + hi * 8 + r;
            Out[(size_t)m * G4H + n] = acc[c][r] + bias;
        }
    }
}

// ---------------------------------------------------------------------------
// Persistent recurrent scan over one chunk of CT steps.
// 32 workgroups x 256 threads. WG `ht` owns h columns [ht*16, ht*16+16).
// Prologue: async-stage this WG's 64KB Wh slice into LDS in fragment order.
// Per step: gates[64, 4x16] = h_prev @ Wh^T (WMMA, B from LDS), + xg, cell
// update. h ping-pongs between two bf16 buffers; one atomic-counter grid
// barrier per step publishes h to all WGs.
// ---------------------------------------------------------------------------
__global__ __launch_bounds__(256) void lstm_scan_kernel(
    const float* __restrict__ xg,        // [CT, 64, 2048]
    const __bf16* __restrict__ Whb,      // [2048, 512] bf16
    __bf16* __restrict__ h0, __bf16* __restrict__ h1,  // [64,512] ping-pong
    float* __restrict__ hf,              // [64,512] f32 (final h for h_n)
    float* __restrict__ c_st,            // [64,512] f32 cell state
    float* __restrict__ out_base,        // [T,64,512] (writes rows t0..t0+CT)
    int t0,
    unsigned* __restrict__ bar_cnt) {
    extern __shared__ char smem[];
    __bf16* whf = (__bf16*)smem;                              // 64 KB
    float (*gates)[BB][16] = (float (*)[BB][16])(smem + WHF_BYTES);

    int lane = threadIdx.x & 31;
    int wave = threadIdx.x >> 5;
    int mt   = wave & 3;         // M-tile (16 rows of batch)
    int gp   = wave >> 2;        // 0 -> gates f,i ; 1 -> gates g,o
    int j0   = blockIdx.x * 16;  // owned h columns
    unsigned nwg = gridDim.x;

    // ---- stage Wh slice into LDS, fragment-ordered, via async copies ----
    // Element e = ((g*16+kt)*32 + L)*16 + j  <-  Whb[g*512+j0+(L&15)]
    //                                             [kt*32 + (L>>4)*16 + j]
    for (int s = threadIdx.x; s < 4096; s += 256) {   // 16B segments
        int jseg = (s & 1) * 8;
        int f    = s >> 1;          // (g*16+kt)*32 + L
        int L    = f & 31;
        int kt   = (f >> 5) & 15;
        int g    = f >> 9;
        const __bf16* src = Whb + (size_t)(g * HH + j0 + (L & 15)) * HH +
                            kt * 32 + (L >> 4) * 16 + jseg;
        async_load_b128((unsigned)(s * 16), (const void*)src);
    }
    wait_asynccnt0();
    __syncthreads();

    for (int tl = 0; tl < CT; ++tl) {
        __bf16* hr = (tl & 1) ? h1 : h0;   // read previous h
        __bf16* hw = (tl & 1) ? h0 : h1;   // write new h

        v8f acc0 = v8f{}, acc1 = v8f{};
        for (int k0 = 0; k0 < HH; k0 += 32) {
            int kt = k0 >> 5;
            v16bf a  = load_a_frag_bf16(hr, HH, mt * 16, k0, lane);
            v16bf b0 = load_b_frag_sm(whf, 2 * gp + 0, kt, lane);
            v16bf b1 = load_b_frag_sm(whf, 2 * gp + 1, kt, lane);
            acc0 = wmma_bf16(a, b0, acc0);
            acc1 = wmma_bf16(a, b1, acc1);
        }
        int hi = lane >> 4, jc = lane & 15;
#pragma unroll
        for (int r = 0; r < 8; ++r) {
            int m = mt * 16 + hi * 8 + r;
            gates[2 * gp + 0][m][jc] = acc0[r];
            gates[2 * gp + 1][m][jc] = acc1[r];
        }
        __syncthreads();

        const float* xgt = xg + (size_t)tl * BB * G4H;
        for (int e = threadIdx.x; e < BB * 16; e += 256) {
            int b = e >> 4, j = e & 15;
            int col = j0 + j;
            float fv = gates[0][b][j] + xgt[(size_t)b * G4H + 0 * HH + col];
            float iv = gates[1][b][j] + xgt[(size_t)b * G4H + 1 * HH + col];
            float gv = gates[2][b][j] + xgt[(size_t)b * G4H + 2 * HH + col];
            float ov = gates[3][b][j] + xgt[(size_t)b * G4H + 3 * HH + col];
            float ff = 1.f / (1.f + __expf(-fv));
            float ii = 1.f / (1.f + __expf(-iv));
            float gg = tanhf(gv);
            float oo = 1.f / (1.f + __expf(-ov));
            float cc = ff * c_st[(size_t)b * HH + col] + ii * gg;
            float hh = oo * tanhf(cc);
            c_st[(size_t)b * HH + col] = cc;
            hf[(size_t)b * HH + col] = hh;
            hw[(size_t)b * HH + col] = (__bf16)hh;
            out_base[((size_t)(t0 + tl) * BB + b) * HH + col] = hh;
        }

        // device-wide barrier: all WGs must publish hw before next step
        __threadfence();
        __syncthreads();
        if (threadIdx.x == 0) {
            atomicAdd(bar_cnt, 1u);
            unsigned target = (unsigned)(tl + 1) * nwg;
            while (*((volatile unsigned*)bar_cnt) < target) {
                __builtin_amdgcn_s_sleep(1);
            }
        }
        __syncthreads();
        __threadfence();
    }
}

// ---------------------------------------------------------------------------
// Host-side orchestration
// ---------------------------------------------------------------------------
extern "C" void kernel_launch(void* const* d_in, const int* in_sizes, int n_in,
                              void* d_out, int out_size, void* d_ws,
                              size_t ws_size, hipStream_t stream) {
    const float* X  = (const float*)d_in[0];
    const float* Wx[2] = {(const float*)d_in[1], (const float*)d_in[5]};
    const float* Wh[2] = {(const float*)d_in[2], (const float*)d_in[6]};
    const float* bx[2] = {(const float*)d_in[3], (const float*)d_in[7]};
    const float* bh[2] = {(const float*)d_in[4], (const float*)d_in[8]};
    float* out = (float*)d_out;

    char* ws = (char*)d_ws;
    float*    xg  = (float*)(ws + XG_OFF);
    __bf16*   wxb = (__bf16*)(ws + WXB_OFF);
    __bf16*   whb = (__bf16*)(ws + WHB_OFF);
    __bf16*   h0  = (__bf16*)(ws + HB0_OFF);
    __bf16*   h1  = (__bf16*)(ws + HB1_OFF);
    float*    hf  = (float*)(ws + HF_OFF);
    float*    cst = (float*)(ws + C_OFF);
    unsigned* cnt = (unsigned*)(ws + CNT_OFF);

    const int NCH = TT / CT;             // chunks per layer
    const int NW  = G4H * HH;            // weight element count

    for (int layer = 0; layer < 2; ++layer) {
        cvt_bf16_kernel<<<(NW + 255) / 256, 256, 0, stream>>>(Wx[layer], wxb, NW);
        cvt_bf16_kernel<<<(NW + 255) / 256, 256, 0, stream>>>(Wh[layer], whb, NW);
        prep_kernel<<<64, 256, 0, stream>>>(cnt, h0, h1, hf, cst, 1);

        // layer-1 input is X; layer-2 input is outs1, staged in the outs2
        // region of d_out (read per-chunk before being overwritten by outs2)
        const float* Ain = (layer == 0) ? X : out;

        for (int ch = 0; ch < NCH; ++ch) {
            int t0 = ch * CT;
            const float* Achunk = Ain + (size_t)t0 * BB * HH;
            gemm_xg_kernel<<<dim3(CT * BB / 16, 4), 256, 0, stream>>>(
                Achunk, HH, wxb, bx[layer], bh[layer], xg);
            prep_kernel<<<1, 64, 0, stream>>>(cnt, h0, h1, hf, cst, 0);
            lstm_scan_kernel<<<32, 256, SMEM_BYTES, stream>>>(
                xg, whb, h0, h1, hf, cst, out, t0, cnt);
        }
        float* hn_dst = out + (size_t)TT * BB * HH + (size_t)layer * BB * HH;
        float* cn_dst = out + (size_t)TT * BB * HH + (size_t)2 * BB * HH +
                        (size_t)layer * BB * HH;
        hipMemcpyAsync(hn_dst, hf, ST_BYTES, hipMemcpyDeviceToDevice, stream);
        hipMemcpyAsync(cn_dst, cst, ST_BYTES, hipMemcpyDeviceToDevice, stream);
    }
}
